// EMAVectorQuantizer_36086315221019
// MI455X (gfx1250) — matrix operand order repaired and checked by reading, not compiled, not run on hardware
//
#include <hip/hip_runtime.h>
#include <hip/hip_bf16.h>
#include <stdint.h>

// ---------------------------------------------------------------------------
// VQ codebook search for MI455X (gfx1250, wave32).
//   z: [16, 512, 2048] f32, embeddings: [1024, 512] f32
//   out: z_vq [16,512,2048] f32 followed by 1 f32 loss.
// Strategy: bf16 hi/lo split matmul on v_wmma_f32_16x16x32_bf16
// (dot = zh*eh + zh*el + zl*eh, ~2^-16 rel err), fused distance/argmin,
// then gather + frame-mean loss.
// ---------------------------------------------------------------------------

#define B_DIM   16
#define D_DIM   512
#define T_DIM   2048
#define Z_NUM   1024
#define N_FRM   (B_DIM * T_DIM)          // 32768
#define KSTEPS  (D_DIM / 32)             // 16 K-slabs of 32
#define CTILES  (Z_NUM / 16)             // 64 code tiles
#define MT_PER_WG 4                      // 4 frame-tiles (64 frames) per WG
#define WAVES     4                      // 128 threads
#define LDS_A_BYTES (MT_PER_WG * KSTEPS * 2 /*hi,lo*/ * 4 /*jp*/ * 32 /*lane*/ * 8)
// = 131072 bytes; + 256 bytes znorm. Legal: CDNA5 allows up to 320KB LDS/WG.

typedef __bf16 v16bf __attribute__((ext_vector_type(16)));
typedef float  v8f   __attribute__((ext_vector_type(8)));

struct U4x2 { uint4 a, b; };
struct U2x4 { uint2 a, b, c, d; };

__device__ __forceinline__ uint32_t f32bits(float x)   { return __builtin_bit_cast(uint32_t, x); }
__device__ __forceinline__ float    bitsf32(uint32_t b){ return __builtin_bit_cast(float, b); }

// bf16 round-to-nearest-even (returns 16-bit pattern in low bits)
__device__ __forceinline__ uint32_t bf16rne(float x) {
    uint32_t b = f32bits(x);
    return (b + 0x7FFFu + ((b >> 16) & 1u)) >> 16;
}
// hi part = truncation to top 16 bits (exact split: x - hi is exact in f32)
__device__ __forceinline__ uint32_t packhi(float x0, float x1) {
    return (f32bits(x0) >> 16) | (f32bits(x1) & 0xFFFF0000u);
}
__device__ __forceinline__ uint32_t packlo(float x0, float x1) {
    float h0 = bitsf32(f32bits(x0) & 0xFFFF0000u);
    float h1 = bitsf32(f32bits(x1) & 0xFFFF0000u);
    return bf16rne(x0 - h0) | (bf16rne(x1 - h1) << 16);
}

__device__ __forceinline__ v16bf as_v16bf(uint4 a, uint4 b) {
    U4x2 t{a, b};
    return __builtin_bit_cast(v16bf, t);
}
__device__ __forceinline__ v16bf as_v16bf(uint2 a, uint2 b, uint2 c, uint2 d) {
    U2x4 t{a, b, c, d};
    return __builtin_bit_cast(v16bf, t);
}

// LDS A index (in uint2 units): [mt][s][hl][jp][lane]
__device__ __forceinline__ int aIdx(int mt, int s, int hl, int jp, int lane) {
    return ((((mt * KSTEPS + s) * 2 + hl) * 4 + jp) * 32 + lane);
}

// ---------------------------------------------------------------------------
// Prep: pack embeddings -> WMMA B-layout bf16 hi/lo blobs + ||e||^2, zero loss.
// B order used here (KxN, contiguous k, matches A): lane n (cols) holds K=0..15 in dwords j=0..7 of
// its 32B blob; lane n+16 holds K=16..31. Blob dword index:
//   ((c*16 + s)*32 + lane)*8 + j     (c = code tile, s = K-slab)
// ---------------------------------------------------------------------------
__global__ __launch_bounds__(128)
void vq_prep(const float* __restrict__ emb,
             uint32_t* __restrict__ ehp, uint32_t* __restrict__ elp,
             float* __restrict__ enorm, float* __restrict__ loss)
{
    const int n = blockIdx.x;          // code row 0..1023
    const int t = threadIdx.x;         // 128 threads, 4 floats each
    const float4 v = ((const float4*)(emb + (size_t)n * D_DIM))[t];
    float sq = v.x * v.x + v.y * v.y + v.z * v.z + v.w * v.w;

    const int c  = n >> 4;
    const int nl = n & 15;
#pragma unroll
    for (int p = 0; p < 2; ++p) {
        const int d  = 4 * t + 2 * p;
        const float x0 = p ? v.z : v.x;
        const float x1 = p ? v.w : v.y;
        const int s    = d >> 5;
        const int kk   = d & 31;
        const int lane = nl + (kk & 16);
        const int j    = (kk & 15) >> 1;
        const size_t idx = ((size_t)(c * KSTEPS + s) * 32 + lane) * 8 + j;
        ehp[idx] = packhi(x0, x1);
        elp[idx] = packlo(x0, x1);
    }

    // reduce ||e||^2 across 128 threads (4 waves of 32)
#pragma unroll
    for (int dlt = 1; dlt < 32; dlt <<= 1) sq += __shfl_xor(sq, dlt, 32);
    __shared__ float red[WAVES];
    if ((t & 31) == 0) red[t >> 5] = sq;
    __syncthreads();
    if (t == 0) enorm[n] = red[0] + red[1] + red[2] + red[3];
    if (n == 0 && t == 0) loss[0] = 0.0f;   // re-zeroed every launch (deterministic)
}

// ---------------------------------------------------------------------------
// Main: each WG = 4 waves, 64 frames. Wave w converts frame-tile w of z into
// LDS (bf16 hi/lo, WMMA A-layout, dword-pair SoA -> conflict-free ds_b64),
// then the 4 waves split the 64 code tiles; 3 bf16 WMMAs per K-slab per tile.
// ---------------------------------------------------------------------------
__global__ __launch_bounds__(128)
void vq_main(const float* __restrict__ z, const float* __restrict__ emb,
             const uint32_t* __restrict__ ehp, const uint32_t* __restrict__ elp,
             const float* __restrict__ enorm,
             float* __restrict__ out, float* __restrict__ loss)
{
    extern __shared__ char smem[];
    uint2* aBuf   = (uint2*)smem;                       // 131072 B, A staging
    float* znormS = (float*)(smem + LDS_A_BYTES);       // 64 floats
    // reuse of the (dead) A region after the main loop:
    float*    wMin  = (float*)smem;                     // [4][64]
    uint32_t* wIdx  = (uint32_t*)(smem + 1024);         // [4][64]
    uint32_t* idxF  = (uint32_t*)(smem + 2048);         // [64]
    float*    lossS = (float*)(smem + 2304);

    const int tid  = threadIdx.x;
    const int w    = tid >> 5;
    const int lane = tid & 31;
    const int f0   = blockIdx.x * 64;       // first frame of this WG
    const int b0   = f0 >> 11;              // / T_DIM
    const int t0   = f0 & (T_DIM - 1);

    // ---- Phase 1: wave w converts frame-tile mt = w into LDS A-layout ----
    {
        const int mt = w;
        const int m  = lane & 15;           // frame within tile (A row)
        const int kg = lane >> 4;           // K group (0: K 0-15, 1: 16-31)
        const float* zb = z + (size_t)b0 * D_DIM * T_DIM + (t0 + mt * 16 + m);
        float sq = 0.0f;
        for (int s = 0; s < KSTEPS; ++s) {
#pragma unroll
            for (int jp = 0; jp < 4; ++jp) {
                // the 4 consecutive d values feeding dwords j=2jp,2jp+1
                const int dstart = s * 32 + 16 * kg + 4 * jp;
                const float x0 = zb[(size_t)(dstart + 0) * T_DIM];
                const float x1 = zb[(size_t)(dstart + 1) * T_DIM];
                const float x2 = zb[(size_t)(dstart + 2) * T_DIM];
                const float x3 = zb[(size_t)(dstart + 3) * T_DIM];
                sq += x0 * x0 + x1 * x1 + x2 * x2 + x3 * x3;
                uint2 hi, lo;
                hi.x = packhi(x0, x1); hi.y = packhi(x2, x3);
                lo.x = packlo(x0, x1); lo.y = packlo(x2, x3);
                aBuf[aIdx(mt, s, 0, jp, lane)] = hi;
                aBuf[aIdx(mt, s, 1, jp, lane)] = lo;
            }
        }
        sq += __shfl_xor(sq, 16, 32);       // combine the two K halves of frame m
        if (lane < 16) znormS[mt * 16 + lane] = sq;
    }
    __syncthreads();

    // ---- Phase 2: wave w handles code tiles c = 4*ct + w ----
    float    minv[MT_PER_WG][8];
    uint32_t mini[MT_PER_WG][8];
#pragma unroll
    for (int mt = 0; mt < MT_PER_WG; ++mt)
#pragma unroll
        for (int i = 0; i < 8; ++i) { minv[mt][i] = 3.402823466e38f; mini[mt][i] = 0u; }

#pragma unroll 1
    for (int ct = 0; ct < CTILES / WAVES; ++ct) {
        const int c = ct * WAVES + w;
        v8f acc[MT_PER_WG];
#pragma unroll
        for (int mt = 0; mt < MT_PER_WG; ++mt) acc[mt] = v8f{0,0,0,0,0,0,0,0};

        for (int s = 0; s < KSTEPS; ++s) {
            const size_t bi = ((size_t)(c * KSTEPS + s) * 32 + lane) * 2; // uint4 units
            const uint4* eh4 = (const uint4*)ehp;
            const uint4* el4 = (const uint4*)elp;
            const v16bf bh = as_v16bf(eh4[bi], eh4[bi + 1]);
            const v16bf bl = as_v16bf(el4[bi], el4[bi + 1]);
#pragma unroll
            for (int mt = 0; mt < MT_PER_WG; ++mt) {
                const uint2* ah2 = aBuf + aIdx(mt, s, 0, 0, lane);
                const uint2* al2 = aBuf + aIdx(mt, s, 1, 0, lane);
                const v16bf ah = as_v16bf(ah2[0], ah2[32], ah2[64], ah2[96]);
                const v16bf al = as_v16bf(al2[0], al2[32], al2[64], al2[96]);
                acc[mt] = __builtin_amdgcn_wmma_f32_16x16x32_bf16(
                              false, ah, false, bh, (short)0, acc[mt], false, false);
                acc[mt] = __builtin_amdgcn_wmma_f32_16x16x32_bf16(
                              false, ah, false, bl, (short)0, acc[mt], false, false);
                acc[mt] = __builtin_amdgcn_wmma_f32_16x16x32_bf16(
                              false, al, false, bh, (short)0, acc[mt], false, false);
            }
        }
        // distance s = ||e||^2 - 2*dot ; running per-lane argmin
        const uint32_t ncode = (uint32_t)(c * 16 + (lane & 15));
        const float en = enorm[ncode];
#pragma unroll
        for (int mt = 0; mt < MT_PER_WG; ++mt)
#pragma unroll
            for (int i = 0; i < 8; ++i) {
                const float sv = en - 2.0f * acc[mt][i];
                const bool better = sv < minv[mt][i];
                minv[mt][i] = better ? sv : minv[mt][i];
                mini[mt][i] = better ? ncode : mini[mt][i];
            }
    }

    // ---- Phase 3: cross-lane argmin reduce (16 lanes per half-wave) ----
#pragma unroll
    for (int mt = 0; mt < MT_PER_WG; ++mt)
#pragma unroll
        for (int i = 0; i < 8; ++i) {
            float v = minv[mt][i]; uint32_t id = mini[mt][i];
#pragma unroll
            for (int dlt = 1; dlt < 16; dlt <<= 1) {
                const float    ov = __shfl_xor(v, dlt, 32);
                const uint32_t oi = (uint32_t)__shfl_xor((int)id, dlt, 32);
                if (ov < v || (ov == v && oi < id)) { v = ov; id = oi; }
            }
            minv[mt][i] = v; mini[mt][i] = id;
        }
    __syncthreads();   // all waves done reading aBuf; safe to reuse region

    if ((lane & 15) == 0) {
#pragma unroll
        for (int mt = 0; mt < MT_PER_WG; ++mt)
#pragma unroll
            for (int i = 0; i < 8; ++i) {
                const int fr = mt * 16 + i + ((lane >> 4) << 3);
                wMin[w * 64 + fr] = minv[mt][i];
                wIdx[w * 64 + fr] = mini[mt][i];
            }
    }
    if (tid == 0) *lossS = 0.0f;
    __syncthreads();

    // ---- Phase 4: combine across the 4 waves; loss partial ----
    if (tid < 64) {
        float bv = wMin[tid]; uint32_t bi = wIdx[tid];
#pragma unroll
        for (int w2 = 1; w2 < WAVES; ++w2) {
            const float v = wMin[w2 * 64 + tid];
            const uint32_t ii = wIdx[w2 * 64 + tid];
            if (v < bv || (v == bv && ii < bi)) { bv = v; bi = ii; }
        }
        idxF[tid] = bi;
        atomicAdd(lossS, bv + znormS[tid]);   // ||z-e||^2 for this frame
    }
    __syncthreads();
    if (tid == 0) atomicAdd(loss, (*lossS) * (1.0f / (float)N_FRM));

    // ---- Phase 5: gather z_vq (coalesced 256B stores per d row) ----
    for (int e = tid; e < 64 * D_DIM; e += 128) {
        const int fr = e & 63;
        const int d  = e >> 6;
        out[((size_t)(b0 * D_DIM + d)) * T_DIM + (t0 + fr)] =
            emb[(size_t)idxF[fr] * D_DIM + d];
    }
}

// ---------------------------------------------------------------------------
extern "C" void kernel_launch(void* const* d_in, const int* in_sizes, int n_in,
                              void* d_out, int out_size, void* d_ws, size_t ws_size,
                              hipStream_t stream) {
    (void)in_sizes; (void)n_in; (void)out_size; (void)ws_size;
    const float* zp   = (const float*)d_in[0];   // [16,512,2048]
    const float* emb  = (const float*)d_in[1];   // [1024,512]
    float* out  = (float*)d_out;                 // z_vq flat
    float* loss = out + (size_t)B_DIM * D_DIM * T_DIM;

    // workspace: eh packed (1MB) | el packed (1MB) | e_norms (4KB)  (~2.01MB)
    uint32_t* ehp   = (uint32_t*)d_ws;
    uint32_t* elp   = ehp + (size_t)Z_NUM * D_DIM / 2;
    float*    enorm = (float*)(elp + (size_t)Z_NUM * D_DIM / 2);

    vq_prep<<<Z_NUM, 128, 0, stream>>>(emb, ehp, elp, enorm, loss);

    const size_t dynLds = LDS_A_BYTES + 256;     // 131328 B (< 320KB WGP LDS)
    vq_main<<<N_FRM / 64, 128, dynLds, stream>>>(zp, emb, ehp, elp, enorm, out, loss);
}
